// HalfKP_NNUE_2774548873840
// MI455X (gfx1250) — matrix operand order repaired and checked by reading, not compiled, not run on hardware
//
#include <hip/hip_runtime.h>

// ---------------- problem constants ----------------
#define HIDDEN  256
#define IN2     512          // 2*HIDDEN (fc2 fan-in)
#define MLPH    32
#define TABLE   640
#define BATCH   8192
#define KIDX    30

#define ROWS_PER_BLOCK 128
#define THREADS        256   // 8 wave32 waves
#define ROWS_PER_WAVE  16

// ---------------- LDS layout (bank-conflict padded) ----------------
// strides in elements
#define W2_STRIDE 528        // 512 + 16 f16 pad  -> row stride 1056B = 264 dw ≡ 8 banks
#define W3_STRIDE 48         // 32 + 16 f16 pad
#define H_STRIDE  528
#define H2_STRIDE 40         // 32 + 8 f16 pad
#define H3_STRIDE 36         // f32, 32 + 4 pad

#define W2_OFF 0
#define W3_OFF (W2_OFF + MLPH * W2_STRIDE * 2)            // 33792
#define H_OFF  (W3_OFF + MLPH * W3_STRIDE * 2)            // 36864
#define H2_OFF (H_OFF  + ROWS_PER_BLOCK * H_STRIDE * 2)   // 172032
#define H3_OFF (H2_OFF + ROWS_PER_BLOCK * H2_STRIDE * 2)  // 182272
#define SMEM_BYTES (H3_OFF + ROWS_PER_BLOCK * H3_STRIDE * 4) // 200704 (< 320KB WGP LDS)

typedef __attribute__((ext_vector_type(16))) _Float16 v16h;
typedef __attribute__((ext_vector_type(8)))  _Float16 v8h;
typedef __attribute__((ext_vector_type(4)))  _Float16 v4h;
typedef __attribute__((ext_vector_type(8)))  float    v8f;
typedef __attribute__((ext_vector_type(4)))  float    v4f;

union Frag { v16h v; v8h h[2]; };

__device__ __forceinline__ void store_relu4(_Float16* p, v4f v) {
    v4h h;
#pragma unroll
    for (int i = 0; i < 4; ++i) {
        float f = v[i] > 0.0f ? v[i] : 0.0f;
        h[i] = (_Float16)f;
    }
    *(v4h*)p = h;  // ds_store_b64
}

__global__ __launch_bounds__(THREADS)
void nnue_fused_kernel(const int* __restrict__ idx0, const int* __restrict__ idx1,
                       const float* __restrict__ w1,
                       const float* __restrict__ w2, const float* __restrict__ b2,
                       const float* __restrict__ w3, const float* __restrict__ b3,
                       const float* __restrict__ w4, const float* __restrict__ b4,
                       float* __restrict__ out)
{
    extern __shared__ __align__(16) unsigned char smem[];
    _Float16* W2 = (_Float16*)(smem + W2_OFF);
    _Float16* W3 = (_Float16*)(smem + W3_OFF);
    _Float16* H  = (_Float16*)(smem + H_OFF);
    _Float16* H2 = (_Float16*)(smem + H2_OFF);
    float*    H3 = (float*)   (smem + H3_OFF);

    const int tid     = threadIdx.x;
    const int lane    = tid & 31;
    const int wv      = tid >> 5;
    const int l16     = lane & 15;
    const int halfSel = lane >> 4;              // 0: lanes 0-15, 1: lanes 16-31
    const int rowBase = blockIdx.x * ROWS_PER_BLOCK;
    const int mrow    = wv * ROWS_PER_WAVE;     // this wave's local M-tile base

    // ---- Phase 0a: warm L2 with the embedding tables (global_prefetch_b8) ----
    {
        const unsigned tabBytes = 2u * TABLE * HIDDEN * 4u;
        for (unsigned p = (unsigned)tid * 256u; p < tabBytes; p += THREADS * 256u)
            __builtin_prefetch((const char*)w1 + p, 0, 1);
    }

    // ---- Phase 0b: stage fc2_w / fc3_w into LDS as f16 (natural [out][in] layout) ----
    for (int j = tid; j < MLPH * IN2; j += THREADS) {
        int r = j >> 9, c = j & 511;
        W2[r * W2_STRIDE + c] = (_Float16)w2[j];
    }
    for (int j = tid; j < MLPH * MLPH; j += THREADS) {
        int r = j >> 5, c = j & 31;
        W3[r * W3_STRIDE + c] = (_Float16)w3[j];
    }
    __syncthreads();

    // ---- Phase 1: embedding-bag gather-sum. One wave per batch row:
    //      lane l accumulates channels [4l..4l+3] and [128+4l..] of each table.
    //      Per index: 512B fully-coalesced wave-wide float4 gather (L2-resident).
#pragma unroll 1
    for (int i = 0; i < ROWS_PER_WAVE; ++i) {
        const int rl = mrow + i;            // local row 0..127
        const int rg = rowBase + rl;        // global batch row
        v4f a0 = {}, a1 = {}, c0v = {}, c1v = {};
        const int* i0p = idx0 + rg * KIDX;
        const int* i1p = idx1 + rg * KIDX;
#pragma unroll 5
        for (int k = 0; k < KIDX; ++k) {
            const int e0 = i0p[k];
            const int e1 = i1p[k];
            const v4f* s0 = (const v4f*)(w1 + (size_t)e0 * HIDDEN);
            const v4f* s1 = (const v4f*)(w1 + (size_t)(TABLE + e1) * HIDDEN);
            a0  += s0[lane]; a1  += s0[lane + 32];
            c0v += s1[lane]; c1v += s1[lane + 32];
        }
        _Float16* hr = H + rl * H_STRIDE;
        store_relu4(hr + 4 * lane,       a0);   // cols   0..127  : relu(sum0) lo
        store_relu4(hr + 128 + 4 * lane, a1);   // cols 128..255  : relu(sum0) hi
        store_relu4(hr + 256 + 4 * lane, c0v);  // cols 256..383  : relu(sum1) lo
        store_relu4(hr + 384 + 4 * lane, c1v);  // cols 384..511  : relu(sum1) hi
    }
    // No barrier needed: each wave's WMMA A-tile reads only its own 16 rows
    // (same-wave LDS ordering via s_wait_dscnt).

    // ---- Phase 2: FC2 via v_wmma_f32_16x16x32_f16.
    //      D[16x32-strip] = H[16x512] @ W2^T ; two N-tiles, K-loop of 16.
// A frag (16x32 f16): lane(l16) row = mrow+l16; contiguous K-run of 16 at K+16*halfSel.
    // B frag (32x16 f16): lane col N = l16(+16); contiguous K-run of 16 at K+16*halfSel.
    const float bias2_0 = b2[l16];
    const float bias2_1 = b2[l16 + 16];

    v8f acc0 = {}, acc1 = {};
    const _Float16* Abase  = H  + (mrow + l16) * H_STRIDE + 16 * halfSel;
    const _Float16* B0base = W2 + l16        * W2_STRIDE + 16 * halfSel;
    const _Float16* B1base = W2 + (l16 + 16) * W2_STRIDE + 16 * halfSel;

#pragma unroll
    for (int kt = 0; kt < IN2 / 32; ++kt) {
        const int kb = kt * 32;
        Frag A, B0, B1;
        A.h[0]  = *(const v8h*)(Abase  + kb);
        A.h[1]  = *(const v8h*)(Abase  + kb + 8);
        B0.h[0] = *(const v8h*)(B0base + kb);
        B0.h[1] = *(const v8h*)(B0base + kb + 8);
        B1.h[0] = *(const v8h*)(B1base + kb);
        B1.h[1] = *(const v8h*)(B1base + kb + 8);
        acc0 = __builtin_amdgcn_wmma_f32_16x16x32_f16(false, A.v, false, B0.v,
                                                      (short)0, acc0, false, false);
        acc1 = __builtin_amdgcn_wmma_f32_16x16x32_f16(false, A.v, false, B1.v,
                                                      (short)0, acc1, false, false);
    }

    // bias + ReLU, scatter to H2 per the C/D layout: VGPR i, lane -> M=i+8*halfSel, N=l16(+16)
#pragma unroll
    for (int i = 0; i < 8; ++i) {
        const int m = mrow + i + 8 * halfSel;
        float v0 = acc0[i] + bias2_0; v0 = v0 > 0.0f ? v0 : 0.0f;
        float v1 = acc1[i] + bias2_1; v1 = v1 > 0.0f ? v1 : 0.0f;
        H2[m * H2_STRIDE + l16]      = (_Float16)v0;
        H2[m * H2_STRIDE + l16 + 16] = (_Float16)v1;
    }

    // ---- Phase 3: FC3 via WMMA, K=32 (one MAC step per N-tile), bias preloaded into C.
    v8f c0, c1;
    {
        const float b3_0 = b3[l16], b3_1 = b3[l16 + 16];
#pragma unroll
        for (int i = 0; i < 8; ++i) { c0[i] = b3_0; c1[i] = b3_1; }
    }
    {
        Frag A2, B30, B31;
        const _Float16* A2base = H2 + (mrow + l16) * H2_STRIDE + 16 * halfSel;
        A2.h[0] = *(const v8h*)(A2base);
        A2.h[1] = *(const v8h*)(A2base + 8);
        const _Float16* B3b0 = W3 + l16        * W3_STRIDE + 16 * halfSel;
        const _Float16* B3b1 = W3 + (l16 + 16) * W3_STRIDE + 16 * halfSel;
        B30.h[0] = *(const v8h*)(B3b0); B30.h[1] = *(const v8h*)(B3b0 + 8);
        B31.h[0] = *(const v8h*)(B3b1); B31.h[1] = *(const v8h*)(B3b1 + 8);
        c0 = __builtin_amdgcn_wmma_f32_16x16x32_f16(false, A2.v, false, B30.v,
                                                    (short)0, c0, false, false);
        c1 = __builtin_amdgcn_wmma_f32_16x16x32_f16(false, A2.v, false, B31.v,
                                                    (short)0, c1, false, false);
    }
#pragma unroll
    for (int i = 0; i < 8; ++i) {
        const int m = mrow + i + 8 * halfSel;
        float v0 = c0[i] > 0.0f ? c0[i] : 0.0f;
        float v1 = c1[i] > 0.0f ? c1[i] : 0.0f;
        H3[m * H3_STRIDE + l16]      = v0;
        H3[m * H3_STRIDE + l16 + 16] = v1;
    }

    // ---- Phase 4: FC4 (32 -> 1). Lanes 0-15 each reduce one of the wave's rows.
    // (divergent EXEC is fine here: all WMMA is complete)
    if (lane < 16) {
        const int rl = mrow + lane;
        const float* hr = H3 + rl * H3_STRIDE;
        float s = b4[0];
#pragma unroll
        for (int n = 0; n < MLPH; ++n) s += hr[n] * w4[n];
        out[rowBase + rl] = s;
    }
}

extern "C" void kernel_launch(void* const* d_in, const int* in_sizes, int n_in,
                              void* d_out, int out_size, void* d_ws, size_t ws_size,
                              hipStream_t stream) {
    (void)in_sizes; (void)n_in; (void)d_ws; (void)ws_size; (void)out_size;
    const int*   idx0 = (const int*)  d_in[0];
    const int*   idx1 = (const int*)  d_in[1];
    const float* w1   = (const float*)d_in[2];
    const float* w2   = (const float*)d_in[3];
    const float* b2   = (const float*)d_in[4];
    const float* w3   = (const float*)d_in[5];
    const float* b3   = (const float*)d_in[6];
    const float* w4   = (const float*)d_in[7];
    const float* b4   = (const float*)d_in[8];
    float* out = (float*)d_out;

    dim3 grid(BATCH / ROWS_PER_BLOCK);   // 64 blocks of 128 rows
    nnue_fused_kernel<<<grid, THREADS, SMEM_BYTES, stream>>>(
        idx0, idx1, w1, w2, b2, w3, b3, w4, b4, out);
}
